// Encoder_50113678409911
// MI455X (gfx1250) — compile-verified, matrix operand order checked by reading, not run on hardware
//
#include <hip/hip_runtime.h>
#include <hip/hip_bf16.h>

// ---------------------------------------------------------------------------
// GAT(heads=1) + GCN pipeline for MI455X (gfx1250, wave32, WMMA).
// Dense GEMMs use v_wmma_f32_16x16x32_f16 (f16 in, f32 accum).
// Segment ops use L2-resident f32 atomics; segment-max uses flipped-uint
// atomicMax. Workspace requirement ~95 MB.
// ---------------------------------------------------------------------------

typedef __attribute__((ext_vector_type(16))) _Float16 v16h;
typedef __attribute__((ext_vector_type(8)))  _Float16 v8h;
typedef __attribute__((ext_vector_type(8)))  float    v8f;

#define NN      50000
#define NE      800000
#define EN      (NE + NN)          // edges + self loops
#define HIDC    128                // IN_CH == HID == 128
#define OUTC    64
#define NEG_SLOPE 0.2f

// monotone float<->uint mapping so unsigned atomicMax == float max
__device__ __forceinline__ unsigned flipf(float x) {
    unsigned u = __float_as_uint(x);
    return u ^ ((u >> 31) ? 0xFFFFFFFFu : 0x80000000u);
}
__device__ __forceinline__ float unflipf(unsigned k) {
    return __uint_as_float(k ^ ((k >> 31) ? 0x80000000u : 0xFFFFFFFFu));
}

__device__ __forceinline__ void edge_pair(const int* __restrict__ ei, int idx,
                                          int& src, int& dst) {
    if (idx < NE) { src = ei[idx]; dst = ei[NE + idx]; }
    else          { src = dst = idx - NE; }
}

// ---------------------------------------------------------------------------
__global__ void k_init(float* __restrict__ agg, unsigned* __restrict__ m_bits,
                       float* __restrict__ denom, float* __restrict__ deg,
                       float* __restrict__ out, const float* __restrict__ bc) {
    const int stride = gridDim.x * blockDim.x;
    const int t0 = blockIdx.x * blockDim.x + threadIdx.x;
    for (int i = t0; i < NN * HIDC; i += stride) agg[i] = 0.0f;
    for (int i = t0; i < NN; i += stride) {
        m_bits[i] = 0x007FFFFFu;   // flipf(-inf)
        denom[i] = 0.0f;
        deg[i] = 0.0f;
    }
    for (int i = t0; i < NN * OUTC; i += stride) out[i] = bc[i & (OUTC - 1)];
}

// x -> f16, Wg -> f16 transposed [HID(n) x HID(k)], Wc -> f16 transposed [OUT x HID]
__global__ void k_convert(const float* __restrict__ x, const float* __restrict__ Wg,
                          const float* __restrict__ Wc, _Float16* __restrict__ xh,
                          _Float16* __restrict__ WgT, _Float16* __restrict__ WcT) {
    const int stride = gridDim.x * blockDim.x;
    const int t0 = blockIdx.x * blockDim.x + threadIdx.x;
    for (int i = t0; i < NN * HIDC; i += stride) xh[i] = (_Float16)x[i];
    for (int i = t0; i < HIDC * HIDC; i += stride) {
        int n = i >> 7, k = i & 127;
        WgT[i] = (_Float16)Wg[k * HIDC + n];
    }
    for (int i = t0; i < OUTC * HIDC; i += stride) {
        int n = i >> 7, k = i & 127;
        WcT[i] = (_Float16)Wc[k * OUTC + n];
    }
}

// ---------------------------------------------------------------------------
// h[N,128] = x[N,128] @ Wg[128,128].  8 waves/block: wave w -> col tile w.
__global__ void __launch_bounds__(256)
k_gat_gemm(const _Float16* __restrict__ xh, const _Float16* __restrict__ WgT,
           float* __restrict__ h) {
    const int wave = threadIdx.x >> 5;
    const int lane = threadIdx.x & 31;
    const int rowBase = blockIdx.x * 16;         // N=50000 = 3125 * 16 exactly
    const int colBase = wave * 16;
    const int rsel = lane & 15;
    const int koff = (lane >> 4) * 8;

    const _Float16* arow = xh  + (size_t)(rowBase + rsel) * HIDC;
    const _Float16* brow = WgT + (size_t)(colBase + rsel) * HIDC;

    v8f c = {};
#pragma unroll
    for (int k0 = 0; k0 < HIDC; k0 += 32) {
        union { v16h v; v8h p[2]; } a, b;
        a.p[0] = *(const v8h*)(arow + k0 + koff);
        a.p[1] = *(const v8h*)(arow + k0 + 16 + koff);
        b.p[0] = *(const v8h*)(brow + k0 + koff);
        b.p[1] = *(const v8h*)(brow + k0 + 16 + koff);
        c = __builtin_amdgcn_wmma_f32_16x16x32_f16(false, a.v, false, b.v,
                                                   (short)0, c, false, false);
    }
    const int nIdx = colBase + (lane & 15);
    const int mBase = rowBase + ((lane >> 4) << 3);
#pragma unroll
    for (int i = 0; i < 8; ++i)
        h[(size_t)(mBase + i) * HIDC + nIdx] = c[i];
}

// s = h @ att_src, d = h @ att_dst (one thread per node; h is L2-resident)
__global__ void k_sd(const float* __restrict__ h, const float* __restrict__ att_src,
                     const float* __restrict__ att_dst, float* __restrict__ s,
                     float* __restrict__ d) {
    const int t = blockIdx.x * blockDim.x + threadIdx.x;
    if (t >= NN) return;
    const float* hr = h + (size_t)t * HIDC;
    float ss = 0.0f, dd = 0.0f;
#pragma unroll 4
    for (int cch = 0; cch < HIDC; ++cch) {
        float v = hr[cch];
        ss += v * att_src[cch];
        dd += v * att_dst[cch];
    }
    s[t] = ss;
    d[t] = dd;
}

// per-edge leaky_relu score + segment max over dst
__global__ void k_edge_max(const int* __restrict__ ei, const float* __restrict__ s,
                           const float* __restrict__ d, float* __restrict__ e_buf,
                           unsigned* __restrict__ m_bits) {
    const int idx = blockIdx.x * blockDim.x + threadIdx.x;
    if (idx >= EN) return;
    int src, dst;
    edge_pair(ei, idx, src, dst);
    float e = s[src] + d[dst];
    e = (e > 0.0f) ? e : NEG_SLOPE * e;
    e_buf[idx] = e;
    atomicMax(m_bits + dst, flipf(e));
}

// exp(e - m[dst]); segment-sum denom; degree count
__global__ void k_edge_exp(const int* __restrict__ ei, float* __restrict__ e_buf,
                           const unsigned* __restrict__ m_bits,
                           float* __restrict__ denom, float* __restrict__ deg) {
    const int idx = blockIdx.x * blockDim.x + threadIdx.x;
    if (idx >= EN) return;
    int src, dst;
    edge_pair(ei, idx, src, dst);
    float ex = __expf(e_buf[idx] - unflipf(m_bits[dst]));
    e_buf[idx] = ex;
    atomicAdd(denom + dst, ex);
    atomicAdd(deg + dst, 1.0f);
}

// agg[dst] += alpha * h[src] : one wave32 per edge, 4 channels/lane (float4)
__global__ void __launch_bounds__(256)
k_edge_agg(const int* __restrict__ ei, const float* __restrict__ e_buf,
           const float* __restrict__ denom, const float* __restrict__ h,
           float* __restrict__ agg) {
    const int gwave = (int)((blockIdx.x * blockDim.x + threadIdx.x) >> 5);
    const int lane = threadIdx.x & 31;
    if (gwave >= EN) return;
    int src, dst;
    edge_pair(ei, gwave, src, dst);
    const float alpha = e_buf[gwave] / denom[dst];
    const float4 hv = *(const float4*)(h + (size_t)src * HIDC + lane * 4);
    float* ap = agg + (size_t)dst * HIDC + lane * 4;
    atomicAdd(ap + 0, alpha * hv.x);
    atomicAdd(ap + 1, alpha * hv.y);
    atomicAdd(ap + 2, alpha * hv.z);
    atomicAdd(ap + 3, alpha * hv.w);
}

// x1 = relu(agg + bg) -> f16; dinv = rsqrt(deg)
__global__ void k_relu(const float* __restrict__ agg, const float* __restrict__ bg,
                       _Float16* __restrict__ x1h, const float* __restrict__ deg,
                       float* __restrict__ dinv) {
    const int stride = gridDim.x * blockDim.x;
    const int t0 = blockIdx.x * blockDim.x + threadIdx.x;
    for (int i = t0; i < NN * HIDC; i += stride) {
        float v = agg[i] + bg[i & 127];
        x1h[i] = (_Float16)(v > 0.0f ? v : 0.0f);
    }
    for (int i = t0; i < NN; i += stride) dinv[i] = rsqrtf(deg[i]);
}

// h2[N,64] = x1 @ Wc.  8 waves/block: 2 row tiles x 4 col tiles -> 32 rows.
__global__ void __launch_bounds__(256)
k_gcn_gemm(const _Float16* __restrict__ x1h, const _Float16* __restrict__ WcT,
           float* __restrict__ h2) {
    const int wave = threadIdx.x >> 5;
    const int lane = threadIdx.x & 31;
    const int rowBase = blockIdx.x * 32 + (wave >> 2) * 16;
    const int colBase = (wave & 3) * 16;
    const int rsel = lane & 15;
    const int koff = (lane >> 4) * 8;

    int r = rowBase + rsel;
    if (r >= NN) r = NN - 1;                     // clamp (keeps EXEC full)
    const _Float16* arow = x1h + (size_t)r * HIDC;
    const _Float16* brow = WcT + (size_t)(colBase + rsel) * HIDC;

    v8f c = {};
#pragma unroll
    for (int k0 = 0; k0 < HIDC; k0 += 32) {
        union { v16h v; v8h p[2]; } a, b;
        a.p[0] = *(const v8h*)(arow + k0 + koff);
        a.p[1] = *(const v8h*)(arow + k0 + 16 + koff);
        b.p[0] = *(const v8h*)(brow + k0 + koff);
        b.p[1] = *(const v8h*)(brow + k0 + 16 + koff);
        c = __builtin_amdgcn_wmma_f32_16x16x32_f16(false, a.v, false, b.v,
                                                   (short)0, c, false, false);
    }
    const int nIdx = colBase + (lane & 15);
    const int mBase = rowBase + ((lane >> 4) << 3);
#pragma unroll
    for (int i = 0; i < 8; ++i) {
        const int m = mBase + i;
        if (m < NN) h2[(size_t)m * OUTC + nIdx] = c[i];
    }
}

// out[dst] += dinv[src]*dinv[dst] * h2[src] : one wave/edge, 2 ch/lane
__global__ void __launch_bounds__(256)
k_edge_gcn(const int* __restrict__ ei, const float* __restrict__ dinv,
           const float* __restrict__ h2, float* __restrict__ out) {
    const int gwave = (int)((blockIdx.x * blockDim.x + threadIdx.x) >> 5);
    const int lane = threadIdx.x & 31;
    if (gwave >= EN) return;
    int src, dst;
    edge_pair(ei, gwave, src, dst);
    const float w = dinv[src] * dinv[dst];
    const float2 hv = *(const float2*)(h2 + (size_t)src * OUTC + lane * 2);
    float* op = out + (size_t)dst * OUTC + lane * 2;
    atomicAdd(op + 0, w * hv.x);
    atomicAdd(op + 1, w * hv.y);
}

// ---------------------------------------------------------------------------
extern "C" void kernel_launch(void* const* d_in, const int* in_sizes, int n_in,
                              void* d_out, int out_size, void* d_ws, size_t ws_size,
                              hipStream_t stream) {
    (void)in_sizes; (void)n_in; (void)out_size; (void)ws_size;

    const float* x       = (const float*)d_in[0];
    const int*   ei      = (const int*)  d_in[1];
    const float* Wg      = (const float*)d_in[2];
    const float* att_src = (const float*)d_in[3];
    const float* att_dst = (const float*)d_in[4];
    const float* bg      = (const float*)d_in[5];
    const float* Wc      = (const float*)d_in[6];
    const float* bc      = (const float*)d_in[7];
    float* out = (float*)d_out;

    // workspace bump allocator (256B aligned); total ~95 MB
    char* w = (char*)d_ws;
    auto alloc = [&](size_t bytes) -> void* {
        void* p = (void*)w;
        w += (bytes + 255) & ~(size_t)255;
        return p;
    };
    _Float16* xh    = (_Float16*)alloc((size_t)NN * HIDC * 2);
    _Float16* WgT   = (_Float16*)alloc((size_t)HIDC * HIDC * 2);
    _Float16* WcT   = (_Float16*)alloc((size_t)OUTC * HIDC * 2);
    float*    h     = (float*)   alloc((size_t)NN * HIDC * 4);
    float*    s     = (float*)   alloc((size_t)NN * 4);
    float*    dvec  = (float*)   alloc((size_t)NN * 4);
    unsigned* mbits = (unsigned*)alloc((size_t)NN * 4);
    float*    denom = (float*)   alloc((size_t)NN * 4);
    float*    deg   = (float*)   alloc((size_t)NN * 4);
    float*    dinv  = (float*)   alloc((size_t)NN * 4);
    float*    e_buf = (float*)   alloc((size_t)EN * 4);
    float*    agg   = (float*)   alloc((size_t)NN * HIDC * 4);
    _Float16* x1h   = (_Float16*)alloc((size_t)NN * HIDC * 2);
    float*    h2    = (float*)   alloc((size_t)NN * OUTC * 4);

    const dim3 B(256);

    k_init    <<<dim3(1024), B, 0, stream>>>(agg, mbits, denom, deg, out, bc);
    k_convert <<<dim3(1024), B, 0, stream>>>(x, Wg, Wc, xh, WgT, WcT);
    k_gat_gemm<<<dim3(NN / 16), B, 0, stream>>>(xh, WgT, h);
    k_sd      <<<dim3((NN + 255) / 256), B, 0, stream>>>(h, att_src, att_dst, s, dvec);
    k_edge_max<<<dim3((EN + 255) / 256), B, 0, stream>>>(ei, s, dvec, e_buf, mbits);
    k_edge_exp<<<dim3((EN + 255) / 256), B, 0, stream>>>(ei, e_buf, mbits, denom, deg);
    k_edge_agg<<<dim3(EN / 8), B, 0, stream>>>(ei, e_buf, denom, h, agg);
    k_relu    <<<dim3(1024), B, 0, stream>>>(agg, bg, x1h, deg, dinv);
    k_gcn_gemm<<<dim3((NN + 31) / 32), B, 0, stream>>>(x1h, WcT, h2);
    k_edge_gcn<<<dim3(EN / 8), B, 0, stream>>>(ei, dinv, h2, out);
}
